// DecayTemporalGraphNetwork_69063074120440
// MI455X (gfx1250) — matrix operand order repaired and checked by reading, not compiled, not run on hardware
//
#include <hip/hip_runtime.h>
#include <hip/hip_bf16.h>

#define NUM_NODES 1000000
#define MEMW 128
#define NODE_F 172
#define EDGE_F 172
#define KPAD_F 192      // 172 padded to multiple of 32
#define EMB 128
#define BB 16384
#define DECAYF 0.1f

typedef __attribute__((ext_vector_type(16))) __bf16    v16bf;
typedef __attribute__((ext_vector_type(8)))  float     v8f;
typedef __attribute__((ext_vector_type(8)))  unsigned  v8u;   // 32B = one B fragment

static __device__ __forceinline__ int clampid(int id) {
    return id < 0 ? 0 : (id > NUM_NODES - 1 ? NUM_NODES - 1 : id);
}

// ---------------------------------------------------------------------------
// Weight pre-pack: W(k,n)=W[k*sK+n*sN] -> bf16 fragments in WMMA B lane order.
// Fragment (kt,nt): elem j of lane l is W(kt*32 + (l>=16?8:0) + j + (j>=8?8:0), nt*16 + (l&15)).
// Stored contiguously so each lane's 16 elems are one 32-byte load.
// ---------------------------------------------------------------------------
__global__ void tgn_pack_w(const float* __restrict__ W, int sK, int sN,
                           int K, int N, int Kp, __bf16* __restrict__ out)
{
    int idx = blockIdx.x * blockDim.x + threadIdx.x;
    if (idx >= Kp * N) return;
    int j    = idx & 15;
    int lane = (idx >> 4) & 31;
    int frag = idx >> 9;
    int tilesN = N >> 4;
    int kt = frag / tilesN, nt = frag % tilesN;
    int k = (kt << 5) + ((lane >= 16) ? 8 : 0) + j + ((j >= 8) ? 8 : 0);
    int n = (nt << 4) + (lane & 15);
    float v = (k < K) ? W[(size_t)k * sK + (size_t)n * sN] : 0.0f;
    out[idx] = (__bf16)v;
}

// Zero-pad activations [M,K] -> [M,Kp]
__global__ void tgn_pad_a(const float* __restrict__ A, int K, int Kp,
                          float* __restrict__ out, int M)
{
    int idx = blockIdx.x * blockDim.x + threadIdx.x;
    if (idx >= M * Kp) return;
    int m = idx / Kp, c = idx - m * Kp;
    out[idx] = (c < K) ? A[(size_t)m * K + c] : 0.0f;
}

// ---------------------------------------------------------------------------
// bf16-WMMA GEMM, branch-free inner loop. Kp % 32 == 0, N % 32 == 0.
// One wave -> 16x32 output tile (two accumulators sharing one A fragment).
// A: fp32, loaded as 4x float4 per fragment, converted in-register.
// Wp: pre-packed bf16 fragments, one 32B vector load per B fragment.
// ---------------------------------------------------------------------------
__global__ void tgn_gemm(const float* __restrict__ A, int lda,
                         const __bf16* __restrict__ Wp,
                         const float* __restrict__ bias,
                         float* __restrict__ C, int ldc, int coff,
                         int M, int N, int Kp, int act)
{
    const int wave = blockIdx.x * (blockDim.x >> 5) + (threadIdx.x >> 5);
    const int lane = threadIdx.x & 31;
    const int tilesN2 = N >> 5;
    const int tilesM  = M >> 4;
    if (wave >= tilesM * tilesN2) return;          // wave-uniform exit
    const int tm = wave / tilesN2, tn2 = wave % tilesN2;
    const int m0 = tm << 4, n0 = tn2 << 5;

    const int  row = lane & 15;
    const bool hi  = lane >= 16;
    const int  m   = m0 + row;
    const int  abase = hi ? 8 : 0;
    const int  tilesN = N >> 4;
    const int  nt0 = tn2 << 1;

    const float4* __restrict__ arow = (const float4*)(A + (size_t)m * lda);
    const int Kt = Kp >> 5;

    v8f acc0 = {}, acc1 = {};
    for (int kt = 0; kt < Kt; ++kt) {
        const int base4 = ((kt << 5) + abase) >> 2;     // float4 index
        float4 c0 = arow[base4];
        float4 c1 = arow[base4 + 1];
        float4 c2 = arow[base4 + 4];
        float4 c3 = arow[base4 + 5];
        v16bf af;
        af[0] = (__bf16)c0.x;  af[1] = (__bf16)c0.y;  af[2] = (__bf16)c0.z;  af[3] = (__bf16)c0.w;
        af[4] = (__bf16)c1.x;  af[5] = (__bf16)c1.y;  af[6] = (__bf16)c1.z;  af[7] = (__bf16)c1.w;
        af[8] = (__bf16)c2.x;  af[9] = (__bf16)c2.y;  af[10] = (__bf16)c2.z; af[11] = (__bf16)c2.w;
        af[12] = (__bf16)c3.x; af[13] = (__bf16)c3.y; af[14] = (__bf16)c3.z; af[15] = (__bf16)c3.w;

        const size_t fb = ((size_t)(kt * tilesN + nt0) * 32 + lane) * 16;
        v16bf b0 = __builtin_bit_cast(v16bf, *(const v8u*)(Wp + fb));
        v16bf b1 = __builtin_bit_cast(v16bf, *(const v8u*)(Wp + fb + 512));

        acc0 = __builtin_amdgcn_wmma_f32_16x16x32_bf16(
            false, af, false, b0, (short)0, acc0, false, false);
        acc1 = __builtin_amdgcn_wmma_f32_16x16x32_bf16(
            false, af, false, b1, (short)0, acc1, false, false);
    }

    const float bv0 = bias[n0 + row];
    const float bv1 = bias[n0 + 16 + row];
#pragma unroll
    for (int i = 0; i < 8; ++i) {
        int mm = m0 + i + (hi ? 8 : 0);
        float v0 = acc0[i] + bv0;
        float v1 = acc1[i] + bv1;
        if (act) { v0 = fmaxf(v0, 0.0f); v1 = fmaxf(v1, 0.0f); }
        float* crow = C + (size_t)mm * ldc + coff + n0 + row;
        crow[0]  = v0;
        crow[16] = v1;
    }
}

// ---------------------------------------------------------------------------
// Winner maps (last occurrence wins)
// ---------------------------------------------------------------------------
__global__ void tgn_reset_maps(const int* __restrict__ src, const int* __restrict__ dst,
                               int* __restrict__ ms, int* __restrict__ md)
{
    int b = blockIdx.x * blockDim.x + threadIdx.x;
    if (b >= BB) return;
    ms[clampid(src[b])] = -1;
    md[clampid(dst[b])] = -1;
}

__global__ void tgn_scatter_maps(const int* __restrict__ src, const int* __restrict__ dst,
                                 int* __restrict__ ms, int* __restrict__ md)
{
    int b = blockIdx.x * blockDim.x + threadIdx.x;
    if (b >= BB) return;
    atomicMax(&ms[clampid(src[b])], b);
    atomicMax(&md[clampid(dst[b])], b);
}

// Round-1 gather (src): decayed memory0 rows -> H
__global__ void tgn_gather1(const int* __restrict__ ids, const float* __restrict__ ts,
                            const float* __restrict__ mem0, const float* __restrict__ lu0,
                            float* __restrict__ H)
{
    int idx = blockIdx.x * blockDim.x + threadIdx.x;
    if (idx >= BB * MEMW) return;
    int b = idx >> 7, c = idx & 127;
    int id = clampid(ids[b]);
    float dt = fmaxf(ts[b] - lu0[id], 0.0f);
    float s = expf(-DECAYF * dt);
    H[idx] = mem0[(size_t)id * MEMW + c] * s;
}

// Round-2 gather (dst): sees round-1 updates via src winner map
__global__ void tgn_gather2(const int* __restrict__ ids, const float* __restrict__ ts,
                            const float* __restrict__ mem0, const float* __restrict__ lu0,
                            const int* __restrict__ mapS, const float* __restrict__ NEW1,
                            float* __restrict__ H)
{
    int idx = blockIdx.x * blockDim.x + threadIdx.x;
    if (idx >= BB * MEMW) return;
    int b = idx >> 7, c = idx & 127;
    int id = clampid(ids[b]);
    int w1 = mapS[id];
    float lu; const float* mrow;
    if (w1 >= 0) { lu = ts[w1]; mrow = NEW1 + (size_t)w1 * MEMW; }
    else         { lu = lu0[id]; mrow = mem0 + (size_t)id * MEMW; }
    float dt = fmaxf(ts[b] - lu, 0.0f);
    H[idx] = mrow[c] * expf(-DECAYF * dt);
}

// GRU combine: NEW = (1-z)*n + z*h
__global__ void tgn_gru(const float* __restrict__ GI, const float* __restrict__ GH,
                        const float* __restrict__ H, float* __restrict__ NEW)
{
    int idx = blockIdx.x * blockDim.x + threadIdx.x;
    if (idx >= BB * MEMW) return;
    int b = idx >> 7, c = idx & 127;
    const float* gi = GI + (size_t)b * 3 * MEMW;
    const float* gh = GH + (size_t)b * 3 * MEMW;
    float r = 1.0f / (1.0f + expf(-(gi[c] + gh[c])));
    float z = 1.0f / (1.0f + expf(-(gi[MEMW + c] + gh[MEMW + c])));
    float n = tanhf(gi[2 * MEMW + c] + r * gh[2 * MEMW + c]);
    NEW[idx] = (1.0f - z) * n + z * H[idx];
}

// Embedding gather: dst-winner > src-winner > memory0; attended = mem*score^2
__global__ void tgn_gather_emb(const int* __restrict__ ids, const float* __restrict__ ts,
                               const float* __restrict__ mem0, const float* __restrict__ lu0,
                               const int* __restrict__ mapS, const int* __restrict__ mapD,
                               const float* __restrict__ NEW1, const float* __restrict__ NEW2,
                               float* __restrict__ COMB, float* __restrict__ score_out)
{
    int idx = blockIdx.x * blockDim.x + threadIdx.x;
    if (idx >= BB * MEMW) return;
    int b = idx >> 7, c = idx & 127;
    int id = clampid(ids[b]);
    int w2 = mapD[id], w1 = mapS[id];
    float lu; const float* mrow;
    if (w2 >= 0)      { lu = ts[w2]; mrow = NEW2 + (size_t)w2 * MEMW; }
    else if (w1 >= 0) { lu = ts[w1]; mrow = NEW1 + (size_t)w1 * MEMW; }
    else              { lu = lu0[id]; mrow = mem0 + (size_t)id * MEMW; }
    float dt = fmaxf(ts[b] - lu, 0.0f);
    float s = expf(-DECAYF * dt);
    COMB[(size_t)b * 256 + c] = mrow[c] * s * s;
    if (c == 0) score_out[b] = s;
}

// Link head: sigmoid(L1 . W_l2 + b_l2)
__global__ void tgn_link_head(const float* __restrict__ L1, const float* __restrict__ Wl2,
                              const float* __restrict__ bl2, float* __restrict__ out)
{
    int b = blockIdx.x * blockDim.x + threadIdx.x;
    if (b >= BB) return;
    float s = bl2[0];
    const float* r = L1 + (size_t)b * EMB;
#pragma unroll 8
    for (int t = 0; t < EMB; ++t) s += r[t] * Wl2[t];
    out[b] = 1.0f / (1.0f + expf(-s));
}

// ---------------------------------------------------------------------------
// Host-side launcher
// ---------------------------------------------------------------------------
static inline int gemm_blocks(int M, int N) { return ((M >> 4) * (N >> 5)) / 8; }
static inline int ceil_div(int a, int b)    { return (a + b - 1) / b; }

extern "C" void kernel_launch(void* const* d_in, const int* in_sizes, int n_in,
                              void* d_out, int out_size, void* d_ws, size_t ws_size,
                              hipStream_t stream) {
    (void)in_sizes; (void)n_in; (void)out_size; (void)ws_size;

    const int*   src_ids  = (const int*)  d_in[0];
    const int*   dst_ids  = (const int*)  d_in[1];
    const float* src_feat = (const float*)d_in[2];
    const float* dst_feat = (const float*)d_in[3];
    const float* ts       = (const float*)d_in[4];
    const float* edge_feat= (const float*)d_in[5];
    const float* mem0     = (const float*)d_in[6];
    const float* lu0      = (const float*)d_in[7];
    const float* W_edge = (const float*)d_in[8];  const float* b_edge = (const float*)d_in[9];
    const float* W_mp1  = (const float*)d_in[10]; const float* b_mp1  = (const float*)d_in[11];
    const float* W_mp2  = (const float*)d_in[12]; const float* b_mp2  = (const float*)d_in[13];
    const float* W_ih   = (const float*)d_in[14]; const float* W_hh   = (const float*)d_in[15];
    const float* b_ih   = (const float*)d_in[16]; const float* b_hh   = (const float*)d_in[17];
    const float* W_node = (const float*)d_in[18]; const float* b_node = (const float*)d_in[19];
    const float* W_p1   = (const float*)d_in[20]; const float* b_p1   = (const float*)d_in[21];
    const float* W_p2   = (const float*)d_in[22]; const float* b_p2   = (const float*)d_in[23];
    const float* W_l1   = (const float*)d_in[24]; const float* b_l1   = (const float*)d_in[25];
    const float* W_l2   = (const float*)d_in[26]; const float* b_l2   = (const float*)d_in[27];

    float* out_link = (float*)d_out;            // [B]
    float* out_ss   = out_link + BB;            // src_score [B]
    float* out_ds   = out_ss   + BB;            // dst_score [B]

    // workspace carve-up (all offsets are multiples of 32 bytes)
    char* p = (char*)d_ws;
    int*   mapS = (int*)p;   p += (size_t)NUM_NODES * 4;
    int*   mapD = (int*)p;   p += (size_t)NUM_NODES * 4;
    float* A1   = (float*)p; p += (size_t)BB * 128 * 4;   // E / P / T
    float* A2   = (float*)p; p += (size_t)BB * 128 * 4;   // H1 / H / L1
    float* G    = (float*)p; p += (size_t)BB * 384 * 4;   // GI, later LINK [B,256]
    float* G2   = (float*)p; p += (size_t)BB * 384 * 4;   // GH
    float* N1   = (float*)p; p += (size_t)BB * 128 * 4;   // new mem (src round)
    float* N2   = (float*)p; p += (size_t)BB * 128 * 4;   // new mem (dst round)
    float* C1   = (float*)p; p += (size_t)BB * 256 * 4;   // [attended_src | node_src]
    float* C2   = (float*)p; p += (size_t)BB * 256 * 4;   // [attended_dst | node_dst]
    float* EP   = (float*)p; p += (size_t)BB * KPAD_F * 4; // padded edge_features
    float* SP   = (float*)p; p += (size_t)BB * KPAD_F * 4; // padded src_features
    float* DP   = (float*)p; p += (size_t)BB * KPAD_F * 4; // padded dst_features
    __bf16* WpE  = (__bf16*)p; p += (size_t)KPAD_F * 128 * 2;
    __bf16* WpM1 = (__bf16*)p; p += (size_t)128 * 128 * 2;
    __bf16* WpM2 = (__bf16*)p; p += (size_t)128 * 128 * 2;
    __bf16* WpIh = (__bf16*)p; p += (size_t)128 * 384 * 2;
    __bf16* WpHh = (__bf16*)p; p += (size_t)128 * 384 * 2;
    __bf16* WpN  = (__bf16*)p; p += (size_t)KPAD_F * 128 * 2;
    __bf16* WpP1 = (__bf16*)p; p += (size_t)256 * 128 * 2;
    __bf16* WpP2 = (__bf16*)p; p += (size_t)128 * 128 * 2;
    __bf16* WpL1 = (__bf16*)p; p += (size_t)256 * 128 * 2;
    float* LINK = G;                                       // reuse after GI consumed

    const dim3 blk(256);
    const int eb = ceil_div(BB * MEMW, 256);
    const int rb = ceil_div(BB, 256);
    const int pb = ceil_div(BB * KPAD_F, 256);

    // weight packing (bf16, B-fragment layout)  W(k,n) = W[k*sK + n*sN]
    tgn_pack_w<<<ceil_div(KPAD_F*128,256), blk, 0, stream>>>(W_edge, 128, 1, EDGE_F, 128, KPAD_F, WpE);
    tgn_pack_w<<<ceil_div(128*128,256),    blk, 0, stream>>>(W_mp1,  128, 1, 128, 128, 128, WpM1);
    tgn_pack_w<<<ceil_div(128*128,256),    blk, 0, stream>>>(W_mp2,  128, 1, 128, 128, 128, WpM2);
    tgn_pack_w<<<ceil_div(128*384,256),    blk, 0, stream>>>(W_ih,   1, 128, 128, 384, 128, WpIh);  // W_ih^T
    tgn_pack_w<<<ceil_div(128*384,256),    blk, 0, stream>>>(W_hh,   1, 128, 128, 384, 128, WpHh);  // W_hh^T
    tgn_pack_w<<<ceil_div(KPAD_F*128,256), blk, 0, stream>>>(W_node, 128, 1, NODE_F, 128, KPAD_F, WpN);
    tgn_pack_w<<<ceil_div(256*128,256),    blk, 0, stream>>>(W_p1,   128, 1, 256, 128, 256, WpP1);
    tgn_pack_w<<<ceil_div(128*128,256),    blk, 0, stream>>>(W_p2,   128, 1, 128, 128, 128, WpP2);
    tgn_pack_w<<<ceil_div(256*128,256),    blk, 0, stream>>>(W_l1,   128, 1, 256, 128, 256, WpL1);

    // activation padding 172 -> 192
    tgn_pad_a<<<pb, blk, 0, stream>>>(edge_feat, EDGE_F, KPAD_F, EP, BB);
    tgn_pad_a<<<pb, blk, 0, stream>>>(src_feat,  NODE_F, KPAD_F, SP, BB);
    tgn_pad_a<<<pb, blk, 0, stream>>>(dst_feat,  NODE_F, KPAD_F, DP, BB);

    // winner maps
    tgn_reset_maps  <<<rb, blk, 0, stream>>>(src_ids, dst_ids, mapS, mapD);
    tgn_scatter_maps<<<rb, blk, 0, stream>>>(src_ids, dst_ids, mapS, mapD);

    // shared message pipeline:  E -> relu(E@W_mp1) -> P -> GI = P@W_ih^T
    tgn_gemm<<<gemm_blocks(BB,128), blk, 0, stream>>>(EP, KPAD_F, WpE,  b_edge, A1, 128, 0, BB, 128, KPAD_F, 0);
    tgn_gemm<<<gemm_blocks(BB,128), blk, 0, stream>>>(A1, 128,    WpM1, b_mp1,  A2, 128, 0, BB, 128, 128, 1);
    tgn_gemm<<<gemm_blocks(BB,128), blk, 0, stream>>>(A2, 128,    WpM2, b_mp2,  A1, 128, 0, BB, 128, 128, 0);
    tgn_gemm<<<gemm_blocks(BB,384), blk, 0, stream>>>(A1, 128,    WpIh, b_ih,   G,  384, 0, BB, 384, 128, 0);

    // round 1 (src)
    tgn_gather1<<<eb, blk, 0, stream>>>(src_ids, ts, mem0, lu0, A2);
    tgn_gemm<<<gemm_blocks(BB,384), blk, 0, stream>>>(A2, 128, WpHh, b_hh, G2, 384, 0, BB, 384, 128, 0);
    tgn_gru<<<eb, blk, 0, stream>>>(G, G2, A2, N1);

    // round 2 (dst) — sees round-1 updates
    tgn_gather2<<<eb, blk, 0, stream>>>(dst_ids, ts, mem0, lu0, mapS, N1, A2);
    tgn_gemm<<<gemm_blocks(BB,384), blk, 0, stream>>>(A2, 128, WpHh, b_hh, G2, 384, 0, BB, 384, 128, 0);
    tgn_gru<<<eb, blk, 0, stream>>>(G, G2, A2, N2);

    // embeddings: node GEMMs into COMB[:,128:256], attended into COMB[:,0:128]
    tgn_gemm<<<gemm_blocks(BB,128), blk, 0, stream>>>(SP, KPAD_F, WpN, b_node, C1, 256, 128, BB, 128, KPAD_F, 0);
    tgn_gemm<<<gemm_blocks(BB,128), blk, 0, stream>>>(DP, KPAD_F, WpN, b_node, C2, 256, 128, BB, 128, KPAD_F, 0);
    tgn_gather_emb<<<eb, blk, 0, stream>>>(src_ids, ts, mem0, lu0, mapS, mapD, N1, N2, C1, out_ss);
    tgn_gather_emb<<<eb, blk, 0, stream>>>(dst_ids, ts, mem0, lu0, mapS, mapD, N1, N2, C2, out_ds);

    // projection: relu(COMB@W_p1)@W_p2 -> LINK halves  (GI dead, reuse G)
    tgn_gemm<<<gemm_blocks(BB,128), blk, 0, stream>>>(C1, 256, WpP1, b_p1, A1, 128, 0, BB, 128, 256, 1);
    tgn_gemm<<<gemm_blocks(BB,128), blk, 0, stream>>>(A1, 128, WpP2, b_p2, LINK, 256, 0, BB, 128, 128, 0);
    tgn_gemm<<<gemm_blocks(BB,128), blk, 0, stream>>>(C2, 256, WpP1, b_p1, A1, 128, 0, BB, 128, 256, 1);
    tgn_gemm<<<gemm_blocks(BB,128), blk, 0, stream>>>(A1, 128, WpP2, b_p2, LINK, 256, 128, BB, 128, 128, 0);

    // link predictor
    tgn_gemm<<<gemm_blocks(BB,128), blk, 0, stream>>>(LINK, 256, WpL1, b_l1, A2, 128, 0, BB, 128, 256, 1);
    tgn_link_head<<<rb, blk, 0, stream>>>(A2, W_l2, b_l2, out_link);
}
